// Part_Cls_5119601017101
// MI455X (gfx1250) — matrix operand order repaired and checked by reading, not compiled, not run on hardware
//
#include <hip/hip_runtime.h>
#include <hip/hip_bf16.h>
#include <stdint.h>

// ---------------------------------------------------------------------------
// Problem constants (reference): B=16, F=256, P=8192, C=96, H=256
// ---------------------------------------------------------------------------
#define BB     16
#define FF     256
#define PP     8192
#define CC     96
#define HH     256
#define NSEG   (BB * CC)            // 1536 pooled rows
#define CHUNKS 8                    // point chunks per batch
#define PTSC   (PP / CHUNKS)        // 1024 points per chunk
#define PPT    4                    // 4 CONSECUTIVE points per thread (float4)
#define FHALF  (FF / 2)             // process features in 2 halves (LDS budget)
#define TSTR   (FHALF + 1)          // 129: (c*129+f)%64=(c+f)%64 -> no bank clash
#define GK     256                  // K dim of every GEMM layer

typedef __attribute__((ext_vector_type(16))) _Float16 v16h;
typedef __attribute__((ext_vector_type(8)))  _Float16 v8h;
typedef __attribute__((ext_vector_type(8)))  float    v8f;

// Order-preserving float->u32 key: larger float <=> larger key. Key 0 is the
// minimum (corresponds to -NaN, never produced by finite inputs).
__device__ __forceinline__ unsigned f32_key(float v) {
    int b = __float_as_int(v);
    return (unsigned)b ^ (((unsigned)(b >> 31)) | 0x80000000u);
}
__device__ __forceinline__ float key_f32(unsigned k) {
    int b = (k & 0x80000000u) ? (int)(k ^ 0x80000000u) : (int)~k;
    return __int_as_float(b);
}

// ---------------------------------------------------------------------------
// Kernel 0: zero the per-segment counts (accumulated atomically per call)
// ---------------------------------------------------------------------------
__global__ void pc_zero_counts_kernel(int* __restrict__ counts) {
    int i = blockIdx.x * blockDim.x + threadIdx.x;
    if (i < NSEG) counts[i] = 0;
}

// ---------------------------------------------------------------------------
// Kernel 1: chunked segment-max pooling with LDS pre-reduction.
// Grid (CHUNKS, BB), 256 threads. Each thread owns 4 consecutive points ->
// one global_load_b128 per feature (wave covers 512B contiguous). The 33.5M
// max-reductions are single full-EXEC ds_max_u32 ops on monotonic keys into
// a [96][129] tile (bank = (c+f)%64, conflict-free across classes). Partial
// key tiles are spilled with plain coalesced stores -> zero global atomics
// on the hot path.
// ---------------------------------------------------------------------------
__global__ void pc_pool_kernel(const float* __restrict__ feat,
                               const int*   __restrict__ label,
                               unsigned* __restrict__ partial,  // [B][CHUNKS][CC][FF] keys
                               int*      __restrict__ counts) {
    __shared__ unsigned tile[CC * TSTR];
    __shared__ int lcnt[CC];

    const int b   = blockIdx.y;
    const int ch  = blockIdx.x;
    const int tid = threadIdx.x;

    // 4 consecutive points per thread: one int4 label load
    const int pbase = ch * PTSC + tid * PPT;
    const int4 lbv = *(const int4*)(label + b * PP + pbase);
    const int lb[PPT] = {lbv.x, lbv.y, lbv.z, lbv.w};

    if (tid < CC) lcnt[tid] = 0;
    __syncthreads();
#pragma unroll
    for (int i = 0; i < PPT; ++i) atomicAdd(&lcnt[lb[i]], 1);

    const float* __restrict__ src = feat + (size_t)b * FF * PP + pbase;
    unsigned* __restrict__ dst = partial + (size_t)(b * CHUNKS + ch) * CC * FF;

    for (int half = 0; half < 2; ++half) {
        const int f0 = half * FHALF;
        // init tile to minimum key
        for (int j = tid; j < CC * TSTR; j += 256) tile[j] = 0u;
        __syncthreads();

        // reduce this chunk's points into the LDS key tile
#pragma unroll 2
        for (int f = 0; f < FHALF; ++f) {
            const float4 v = *(const float4*)(src + (size_t)(f0 + f) * PP);
            const float vv[PPT] = {v.x, v.y, v.z, v.w};
#pragma unroll
            for (int i = 0; i < PPT; ++i)
                atomicMax(&tile[lb[i] * TSTR + f], f32_key(vv[i]));
        }
        __syncthreads();

        // spill partial key tile (coalesced, no atomics)
        for (int j = tid; j < CC * FHALF; j += 256) {
            const int c = j / FHALF, f = j - c * FHALF;
            dst[(size_t)c * FF + f0 + f] = tile[c * TSTR + f];
        }
        __syncthreads();
    }

    if (tid < CC) atomicAdd(&counts[b * CC + tid], lcnt[tid]);
}

// ---------------------------------------------------------------------------
// Kernel 2: combine the 8 chunk-partials per batch (max of keys), decode,
// apply reference select: counts==0 -> 0 ; counts==P -> segmax ; else relu,
// and emit f16 activations for WMMA.
// ---------------------------------------------------------------------------
__global__ void pc_reduce_kernel(const unsigned* __restrict__ partial,
                                 const int*      __restrict__ counts,
                                 _Float16* __restrict__ a0) {
    const int idx = blockIdx.x * blockDim.x + threadIdx.x;   // NSEG*FF
    const int seg = idx >> 8;                                // /FF
    const int b   = seg / CC;
    const int loc = idx - b * CC * FF;

    const unsigned* __restrict__ pb = partial + (size_t)b * CHUNKS * CC * FF + loc;
    unsigned k = 0u;
#pragma unroll
    for (int ch = 0; ch < CHUNKS; ++ch) {
        const unsigned kv = pb[(size_t)ch * CC * FF];
        k = (kv > k) ? kv : k;
    }

    float m = key_f32(k);
    const int cnt = counts[seg];
    m = (cnt == 0) ? 0.0f : ((cnt == PP) ? m : fmaxf(m, 0.0f));
    a0[idx] = (_Float16)m;
}

// ---------------------------------------------------------------------------
// Kernel 3: f32 -> f16 weight conversion (once per layer, L2-resident)
// ---------------------------------------------------------------------------
__global__ void pc_cvt_kernel(const float* __restrict__ w,
                              _Float16* __restrict__ wh, int n) {
    int i = blockIdx.x * blockDim.x + threadIdx.x;
    if (i < n) wh[i] = (_Float16)w[i];
}

// ---------------------------------------------------------------------------
// Kernel 4: WMMA GEMM  D = relu?(A(f16,Mx256) * Bh(f16,256xN) + bias)
// Block = 8 waves sharing one N-tile: B panel (256x16 f16 = 8KB) is staged
// into LDS in WMMA B-operand order (32B/lane/K-step), then each wave runs a
// fully unrolled K-loop: 2x ds_load_b128 (B) + 2x global 16B loads (A) +
// v_wmma_f32_16x16x32_f16 per step. Bias+ReLU fused in the epilogue.
// ---------------------------------------------------------------------------
template <bool RELU, bool OUT_F16>
__global__ void pc_gemm_wmma_kernel(const _Float16* __restrict__ A,
                                    const _Float16* __restrict__ Bh,
                                    const float*    __restrict__ bias,
                                    void* __restrict__ Out, int M, int N) {
    __shared__ _Float16 ldsb[GK * 16];     // 8KB: [kstep][lane][16]

    const int nt  = blockIdx.y;
    const int tid = threadIdx.x;

    // Stage B panel: thread t fills slot (kstep s = t>>5, lane L = t&31).
    // Slot holds Bh[k, col] for k = s*32 + (L>>4)*16 + i, col = nt*16 + (L&15).
    {
        const int s = tid >> 5, L = tid & 31;
        const int colg = nt * 16 + (L & 15);
        const int kb   = s * 32 + (L >> 4) * 16;
        _Float16* d = &ldsb[(s * 32 + L) * 16];
#pragma unroll
        for (int i = 0; i < 16; ++i) d[i] = Bh[(size_t)(kb + i) * N + colg];
    }
    __syncthreads();

    const int wid  = tid >> 5;
    const int lane = tid & 31;
    const int mt   = blockIdx.x * 8 + wid;
    const int lrow = lane & 15;
    const int khi  = lane >> 4;

    const _Float16* __restrict__ arow = A + (size_t)(mt * 16 + lrow) * GK;
    const int col = nt * 16 + lrow;

    v8f acc = {};
#pragma unroll
    for (int s = 0; s < GK / 32; ++s) {
        // A fragment: 16 contiguous f16 (k = khi*16 + i), matching B staging
        v8h alo = *(const v8h*)(arow + s * 32 + khi * 16);
        v8h ahi = *(const v8h*)(arow + s * 32 + khi * 16 + 8);
        v16h a = __builtin_shufflevector(alo, ahi,
                                         0, 1, 2, 3, 4, 5, 6, 7,
                                         8, 9, 10, 11, 12, 13, 14, 15);
        // B fragment: 32B contiguous from LDS
        const _Float16* bp = &ldsb[(s * 32 + lane) * 16];
        v8h blo = *(const v8h*)(bp);
        v8h bhi = *(const v8h*)(bp + 8);
        v16h bm = __builtin_shufflevector(blo, bhi,
                                          0, 1, 2, 3, 4, 5, 6, 7,
                                          8, 9, 10, 11, 12, 13, 14, 15);
        acc = __builtin_amdgcn_wmma_f32_16x16x32_f16(
            false, a, false, bm, (short)0, acc, false, false);
    }

    // Epilogue: D layout -> VGPR r holds row (khi*8 + r), col (lane&15)
    const float bv = bias[col];
#pragma unroll
    for (int r = 0; r < 8; ++r) {
        const int m = mt * 16 + khi * 8 + r;
        float v = acc[r] + bv;
        if (RELU) v = fmaxf(v, 0.0f);
        if (OUT_F16) ((_Float16*)Out)[(size_t)m * N + col] = (_Float16)v;
        else         ((float*)Out)[(size_t)m * N + col] = v;
    }
}

// ---------------------------------------------------------------------------
// Kernel 5: part_label / part_batch written as floats after part_pred
// ---------------------------------------------------------------------------
__global__ void pc_meta_kernel(float* __restrict__ out_label,
                               float* __restrict__ out_batch) {
    int i = blockIdx.x * blockDim.x + threadIdx.x;   // NSEG threads
    out_label[i] = (float)(i % CC);
    out_batch[i] = (float)(i / CC);
}

// ---------------------------------------------------------------------------
// Launcher
// ---------------------------------------------------------------------------
extern "C" void kernel_launch(void* const* d_in, const int* in_sizes, int n_in,
                              void* d_out, int out_size, void* d_ws, size_t ws_size,
                              hipStream_t stream) {
    (void)in_sizes; (void)n_in; (void)out_size; (void)ws_size;

    const float* seg_feat  = (const float*)d_in[0];
    const int*   seg_label = (const int*)  d_in[1];
    const float* w1        = (const float*)d_in[2];
    const float* b1        = (const float*)d_in[3];
    const float* w2        = (const float*)d_in[4];
    const float* b2        = (const float*)d_in[5];
    const float* w_logit   = (const float*)d_in[6];
    const float* b_logit   = (const float*)d_in[7];
    float* out = (float*)d_out;

    // Workspace carve-up (offsets 256B aligned, ~14.6MB total)
    char* ws = (char*)d_ws;
    unsigned* partial = (unsigned*)(ws);                 // 16*8*96*256 u32 = 12MB
    int*      counts  = (int*)     (ws + 12582912);      // 1536 i32
    _Float16* a0      = (_Float16*)(ws + 12589312);      // 1536*256 f16
    _Float16* h1      = (_Float16*)(ws + 13375744);      // 1536*256 f16
    _Float16* h2      = (_Float16*)(ws + 14162176);      // 1536*256 f16
    _Float16* w1h     = (_Float16*)(ws + 14948608);      // 256*256 f16
    _Float16* w2h     = (_Float16*)(ws + 15079680);      // 256*256 f16
    _Float16* wlh     = (_Float16*)(ws + 15210752);      // 256*96  f16

    // 0) zero counts
    pc_zero_counts_kernel<<<(NSEG + 255) / 256, 256, 0, stream>>>(counts);

    // 1) LDS-chunked segment-max pooling (single pass over the 128MB tensor)
    {
        dim3 grid(CHUNKS, BB);
        pc_pool_kernel<<<grid, 256, 0, stream>>>(seg_feat, seg_label, partial, counts);
    }

    // 2) combine chunk partials + reference select -> f16 activations
    pc_reduce_kernel<<<(NSEG * FF) / 256, 256, 0, stream>>>(partial, counts, a0);

    // 3) weights f32 -> f16 (tiny, L2-resident)
    pc_cvt_kernel<<<(FF * HH + 255) / 256, 256, 0, stream>>>(w1, w1h, FF * HH);
    pc_cvt_kernel<<<(HH * HH + 255) / 256, 256, 0, stream>>>(w2, w2h, HH * HH);
    pc_cvt_kernel<<<(HH * CC + 255) / 256, 256, 0, stream>>>(w_logit, wlh, HH * CC);

    // 4) MLP via WMMA: 8 waves/block share one N-tile's B panel in LDS
    {
        dim3 g1(NSEG / (16 * 8), HH / 16);   // (12, 16)
        pc_gemm_wmma_kernel<true, true><<<g1, 256, 0, stream>>>(
            a0, w1h, b1, (void*)h1, NSEG, HH);
        pc_gemm_wmma_kernel<true, true><<<g1, 256, 0, stream>>>(
            h1, w2h, b2, (void*)h2, NSEG, HH);
        dim3 g3(NSEG / (16 * 8), CC / 16);   // (12, 6)
        pc_gemm_wmma_kernel<false, false><<<g3, 256, 0, stream>>>(
            h2, wlh, b_logit, (void*)out, NSEG, CC);
    }

    // 5) part_label / part_batch
    pc_meta_kernel<<<NSEG / 256, 256, 0, stream>>>(out + NSEG * CC,
                                                   out + NSEG * CC + NSEG);
}
